// GNNbase_9603546874512
// MI455X (gfx1250) — matrix operand order repaired and checked by reading, not compiled, not run on hardware
//
#include <hip/hip_runtime.h>
#include <math.h>

// ---- problem constants (from the reference) ----
constexpr int kB   = 128;
constexpr int kN   = 100;
constexpr int kE   = 3200;
constexpr int kED  = 8;
constexpr int kH   = 4;
constexpr int kHID = 16;
constexpr int kOUT = 32;
constexpr int kNT  = kB * kN;   // 12800 nodes
constexpr int kET  = kB * kE;   // 409600 edges

typedef __attribute__((ext_vector_type(16))) _Float16 v16h;
typedef __attribute__((ext_vector_type(8)))  float    v8f;

union V16H { v16h v; _Float16 e[16]; };
union V8F  { v8f  v; float    e[8];  };

__global__ void k_zero(float* __restrict__ p, long n) {
  long i = (long)blockIdx.x * blockDim.x + threadIdx.x;
  if (i < n) p[i] = 0.0f;
}

// ---------------------------------------------------------------------------
// Edge MLP: per 16-edge tile, three chained 16x16 WMMA GEMMs
// (K=12,16,16 zero-padded to 32).  ReLU is applied in the D layout; the
// relu'd tile goes to LDS, and LayerNorm is computed per-lane from the full
// LDS row (no cross-lane shuffles), fused into the next A-operand build /
// the final atomic scatter into z[dst].
// One wave per tile, 4 waves per block.
// ---------------------------------------------------------------------------
__global__ __launch_bounds__(128) void k_edge_mlp(
    const float* __restrict__ nodes, const int* __restrict__ eidx,
    const float* __restrict__ eattr, const float* __restrict__ emb,
    const float* __restrict__ W1, const float* __restrict__ b1,
    const float* __restrict__ W2, const float* __restrict__ b2,
    const float* __restrict__ W3, const float* __restrict__ b3,
    const float* __restrict__ lng, const float* __restrict__ lnb,
    float* __restrict__ z)
{
  __shared__ float lds[4][16][17];
  const int lane = threadIdx.x & 31;
  const int wv   = threadIdx.x >> 5;
  const int grp  = lane >> 4;     // half-wave select
  const int n    = lane & 15;     // A-row (edge) AND D-col (feature)
  const int tile = blockIdx.x * 4 + wv;

  // gather the 12 inputs of edge row n: [entity_embed(4) | edge_attr(8)]
  const int f = tile * 16 + n;
  const int b = f / kE, e = f - b * kE;
  const int src = eidx[b * 2 * kE + e]      + b * kN;
  const int dst = eidx[b * 2 * kE + kE + e] + b * kN;

  float in[12];
  const int et = (int)nodes[src];
#pragma unroll
  for (int d = 0; d < 4; ++d) in[d]     = emb[et * 4 + d];
#pragma unroll
  for (int d = 0; d < 8; ++d) in[4 + d] = eattr[(long)f * kED + d];

  // per-lane LN gain/bias for the 8 features this lane materializes
  // (k = grp*8 + h, identical set for A-build and final scatter)
  float gk[8], bk[8];
#pragma unroll
  for (int i = 0; i < 8; ++i) {
    int k = grp * 8 + i;
    gk[i] = lng[k];
    bk[i] = lnb[k];
  }

  V16H a, bm;
  V8F  c;

  // ---- stage 1: [16 x 12] @ W1^T[12 x 16] ----
#pragma unroll
  for (int h = 0; h < 16; ++h) {
    int k0 = h;                     // A layout K map, half-wave 0
    int k1 = h + 16;                // half-wave 1
    float f0 = (k0 < 12) ? in[k0] : 0.0f;
    float f1 = (k1 < 12) ? in[k1] : 0.0f;
    a.e[h] = (_Float16)(grp ? f1 : f0);
  }
#pragma unroll
  for (int h = 0; h < 16; ++h)     // B: lanes 0-15 K=0..15, lanes 16-31 zero pad
    bm.e[h] = (_Float16)((!grp && h < 12) ? W1[n * 12 + h] : 0.0f);
  { float bias = b1[n];
#pragma unroll
    for (int r = 0; r < 8; ++r) c.e[r] = bias; }
  c.v = __builtin_amdgcn_wmma_f32_16x16x32_f16(false, a.v, false, bm.v, (short)0, c.v, false, false);

  // relu -> LDS
#pragma unroll
  for (int r = 0; r < 8; ++r) lds[wv][r + 8 * grp][n] = fmaxf(c.e[r], 0.0f);
  __syncthreads();

  // per-lane LN stats over the full row, fused into A for stage 2
  {
    float v[16];
#pragma unroll
    for (int k = 0; k < 16; ++k) v[k] = lds[wv][n][k];
    float mean = 0.0f;
#pragma unroll
    for (int k = 0; k < 16; ++k) mean += v[k];
    mean *= 0.0625f;
    float var = 0.0f;
#pragma unroll
    for (int k = 0; k < 16; ++k) { float dv = v[k] - mean; var += dv * dv; }
    var *= 0.0625f;
    float inv = 1.0f / sqrtf(var + 1e-5f);
#pragma unroll
    for (int h = 0; h < 8; ++h) {
      float val = grp ? v[8 + h] : v[h];
      a.e[h] = (_Float16)((val - mean) * inv * gk[h] + bk[h]);
    }
#pragma unroll
    for (int h = 8; h < 16; ++h) a.e[h] = (_Float16)0.0f;
  }
  __syncthreads();

  // ---- stage 2: @ W2^T[16 x 16] ----
#pragma unroll
  for (int h = 0; h < 16; ++h)
    bm.e[h] = (_Float16)((h < 8) ? W2[n * 16 + grp * 8 + h] : 0.0f);
  { float bias = b2[n];
#pragma unroll
    for (int r = 0; r < 8; ++r) c.e[r] = bias; }
  c.v = __builtin_amdgcn_wmma_f32_16x16x32_f16(false, a.v, false, bm.v, (short)0, c.v, false, false);

#pragma unroll
  for (int r = 0; r < 8; ++r) lds[wv][r + 8 * grp][n] = fmaxf(c.e[r], 0.0f);
  __syncthreads();

  {
    float v[16];
#pragma unroll
    for (int k = 0; k < 16; ++k) v[k] = lds[wv][n][k];
    float mean = 0.0f;
#pragma unroll
    for (int k = 0; k < 16; ++k) mean += v[k];
    mean *= 0.0625f;
    float var = 0.0f;
#pragma unroll
    for (int k = 0; k < 16; ++k) { float dv = v[k] - mean; var += dv * dv; }
    var *= 0.0625f;
    float inv = 1.0f / sqrtf(var + 1e-5f);
#pragma unroll
    for (int h = 0; h < 8; ++h) {
      float val = grp ? v[8 + h] : v[h];
      a.e[h] = (_Float16)((val - mean) * inv * gk[h] + bk[h]);
    }
#pragma unroll
    for (int h = 8; h < 16; ++h) a.e[h] = (_Float16)0.0f;
  }
  __syncthreads();

  // ---- stage 3: @ W3^T[16 x 16] ----
#pragma unroll
  for (int h = 0; h < 16; ++h)
    bm.e[h] = (_Float16)((h < 8) ? W3[n * 16 + grp * 8 + h] : 0.0f);
  { float bias = b3[n];
#pragma unroll
    for (int r = 0; r < 8; ++r) c.e[r] = bias; }
  c.v = __builtin_amdgcn_wmma_f32_16x16x32_f16(false, a.v, false, bm.v, (short)0, c.v, false, false);

#pragma unroll
  for (int r = 0; r < 8; ++r) lds[wv][r + 8 * grp][n] = fmaxf(c.e[r], 0.0f);
  __syncthreads();

  // final LN fused into the atomic scatter: lane owns its own edge's row n,
  // scattering features k = grp*8 + h into z[dst].
  {
    float v[16];
#pragma unroll
    for (int k = 0; k < 16; ++k) v[k] = lds[wv][n][k];
    float mean = 0.0f;
#pragma unroll
    for (int k = 0; k < 16; ++k) mean += v[k];
    mean *= 0.0625f;
    float var = 0.0f;
#pragma unroll
    for (int k = 0; k < 16; ++k) { float dv = v[k] - mean; var += dv * dv; }
    var *= 0.0625f;
    float inv = 1.0f / sqrtf(var + 1e-5f);
    float* zrow = z + (long)dst * kHID + grp * 8;
#pragma unroll
    for (int h = 0; h < 8; ++h) {
      float val = grp ? v[8 + h] : v[h];
      atomicAdd(&zrow[h], (val - mean) * inv * gk[h] + bk[h]);
    }
  }
}

// ---------------------------------------------------------------------------
// q/k/v/skip projections: one wave per 16-node tile; N-tiles sweep the
// concatenated [q | k | v | skip] columns.  All via WMMA 16x16x32 f16.
// ---------------------------------------------------------------------------
__global__ __launch_bounds__(128) void k_qkv(
    const float* __restrict__ x,
    const float* __restrict__ Wq, const float* __restrict__ bq,
    const float* __restrict__ Wk, const float* __restrict__ bk,
    const float* __restrict__ Wv, const float* __restrict__ bv,
    const float* __restrict__ Ws, const float* __restrict__ bs,
    const int HC, const int outd,
    float* __restrict__ qkv, float* __restrict__ skip)
{
  const int lane = threadIdx.x & 31;
  const int wv   = threadIdx.x >> 5;
  const int grp  = lane >> 4;
  const int n    = lane & 15;
  const int tile = blockIdx.x * 4 + wv;
  const int nodeA = tile * 16 + n;

  V16H a;
#pragma unroll
  for (int h = 0; h < 16; ++h) {
    int k0 = h;
    int k1 = h + 16;
    float f0 = (k0 < 16) ? x[nodeA * 16 + k0] : 0.0f;
    float f1 = (k1 < 16) ? x[nodeA * 16 + k1] : 0.0f;
    a.e[h] = (_Float16)(grp ? f1 : f0);
  }

  const int ntiles = (3 * HC + outd) >> 4;  // 13 (g1) or 26 (g2), exact
  for (int t = 0; t < ntiles; ++t) {
    int j = t * 16 + n;                     // logical output column
    const float* W; float bias; int jj;
    if (j < HC)           { W = Wq; jj = j;          bias = bq[jj]; }
    else if (j < 2 * HC)  { W = Wk; jj = j - HC;     bias = bk[jj]; }
    else if (j < 3 * HC)  { W = Wv; jj = j - 2 * HC; bias = bv[jj]; }
    else                  { W = Ws; jj = j - 3 * HC; bias = bs[jj]; }
    V16H bm;
#pragma unroll
    for (int h = 0; h < 16; ++h)
      bm.e[h] = (_Float16)((!grp) ? W[jj * 16 + h] : 0.0f);
    V8F c;
#pragma unroll
    for (int r = 0; r < 8; ++r) c.e[r] = bias;
    c.v = __builtin_amdgcn_wmma_f32_16x16x32_f16(false, a.v, false, bm.v, (short)0, c.v, false, false);
#pragma unroll
    for (int r = 0; r < 8; ++r) {
      int node = tile * 16 + r + 8 * grp;
      if (j < 3 * HC) qkv[(long)node * (3 * HC) + j]        = c.e[r];
      else            skip[(long)node * outd + (j - 3 * HC)] = c.e[r];
    }
  }
}

// ---------------------------------------------------------------------------
// Per-edge attention pass.  Softmax denominator is factored out:
//   out[dst] = (Σ_e e^{l_e} (v[src]+e)) / (Σ_e e^{l_e} + 1e-16)
// which equals the reference exactly, so one edge pass suffices.
// ---------------------------------------------------------------------------
template <int CH>
__global__ __launch_bounds__(256) void k_edge_attn(
    const int* __restrict__ eidx, const float* __restrict__ eattr,
    const float* __restrict__ qkv, const float* __restrict__ We,
    float* __restrict__ denom, float* __restrict__ num)
{
  constexpr int HC = kH * CH;
  int f = blockIdx.x * blockDim.x + threadIdx.x;
  if (f >= kET) return;
  const int b = f / kE, e = f - b * kE;
  const int src = eidx[b * 2 * kE + e]      + b * kN;
  const int dst = eidx[b * 2 * kE + kE + e] + b * kN;

  float ea[kED];
#pragma unroll
  for (int d = 0; d < kED; ++d) ea[d] = eattr[(long)f * kED + d];

  const float scale = 1.0f / sqrtf((float)CH);
  const float* qd = qkv + (long)dst * 3 * HC;
  const float* ks = qkv + (long)src * 3 * HC + HC;
  const float* vs = qkv + (long)src * 3 * HC + 2 * HC;

  for (int h = 0; h < kH; ++h) {
    float eh[CH];
    float logit = 0.0f;
#pragma unroll
    for (int c = 0; c < CH; ++c) {
      float s = 0.0f;
#pragma unroll
      for (int d = 0; d < kED; ++d) s += ea[d] * We[(h * CH + c) * kED + d];
      eh[c] = s;
      logit += qd[h * CH + c] * (ks[h * CH + c] + s);
    }
    float ex = expf(logit * scale);
    atomicAdd(&denom[(long)dst * kH + h], ex);
#pragma unroll
    for (int c = 0; c < CH; ++c)
      atomicAdd(&num[((long)dst * kH + h) * CH + c], ex * (vs[h * CH + c] + eh[c]));
  }
}

template <int CH>
__global__ __launch_bounds__(256) void k_finalize(
    const float* __restrict__ num, const float* __restrict__ denom,
    const float* __restrict__ skip, float* __restrict__ out)
{
  long id = (long)blockIdx.x * blockDim.x + threadIdx.x;
  if (id >= (long)kNT * CH) return;
  int n = (int)(id / CH), c = (int)(id - (long)n * CH);
  float acc = 0.0f;
#pragma unroll
  for (int h = 0; h < kH; ++h)
    acc += num[((long)n * kH + h) * CH + c] / (denom[(long)n * kH + h] + 1e-16f);
  out[id] = acc * (1.0f / kH) + skip[id];
}

__global__ void k_gather(const float* __restrict__ a2, const int* __restrict__ agent,
                         float* __restrict__ out)
{
  int t = blockIdx.x * blockDim.x + threadIdx.x;
  if (t >= kB * kOUT) return;
  int b = t / kOUT, c = t - b * kOUT;
  int node = b * kN + agent[b];
  out[t] = fmaxf(a2[(long)node * kOUT + c], 0.0f);
}

// ---------------------------------------------------------------------------
extern "C" void kernel_launch(void* const* d_in, const int* in_sizes, int n_in,
                              void* d_out, int out_size, void* d_ws, size_t ws_size,
                              hipStream_t stream) {
  (void)in_sizes; (void)n_in; (void)out_size; (void)ws_size;
  const float* nodes = (const float*)d_in[0];
  const int*   eidx  = (const int*)  d_in[1];
  const float* eattr = (const float*)d_in[2];
  const int*   agent = (const int*)  d_in[3];
  const float* emb   = (const float*)d_in[4];
  const float* W1 = (const float*)d_in[5],  *b1 = (const float*)d_in[6];
  const float* W2 = (const float*)d_in[7],  *b2 = (const float*)d_in[8];
  const float* W3 = (const float*)d_in[9],  *b3 = (const float*)d_in[10];
  const float* lng = (const float*)d_in[11], *lnb = (const float*)d_in[12];
  const float* g1Wq = (const float*)d_in[13], *g1bq = (const float*)d_in[14];
  const float* g1Wk = (const float*)d_in[15], *g1bk = (const float*)d_in[16];
  const float* g1Wv = (const float*)d_in[17], *g1bv = (const float*)d_in[18];
  const float* g1We = (const float*)d_in[19];
  const float* g1Ws = (const float*)d_in[20], *g1bs = (const float*)d_in[21];
  const float* g2Wq = (const float*)d_in[22], *g2bq = (const float*)d_in[23];
  const float* g2Wk = (const float*)d_in[24], *g2bk = (const float*)d_in[25];
  const float* g2Wv = (const float*)d_in[26], *g2bv = (const float*)d_in[27];
  const float* g2We = (const float*)d_in[28];
  const float* g2Ws = (const float*)d_in[29], *g2bs = (const float*)d_in[30];

  // workspace layout (floats)
  float* ws    = (float*)d_ws;
  float* z     = ws;                 // 204800
  float* a1    = ws + 204800;        // 204800
  float* a2    = ws + 409600;        // 409600
  float* qkv   = ws + 819200;        // 12800*384 = 4915200 (reused g1/g2)
  float* skip  = ws + 5734400;       // 12800*32  = 409600  (reused)
  float* denom = ws + 6144000;       // 12800*4   = 51200   (reused)
  float* num   = ws + 6195200;       // 12800*4*32= 1638400 (reused)
  const long accLen = 51200 + 1638400;   // denom+num contiguous

  // ---- embedding edge MLP -> z ----
  k_zero<<<800, 256, 0, stream>>>(z, (long)kNT * kHID);
  k_edge_mlp<<<kET / 16 / 4, 128, 0, stream>>>(nodes, eidx, eattr, emb,
      W1, b1, W2, b2, W3, b3, lng, lnb, z);

  // ---- TransformerConv g1 (ch=16, HC=64) ----
  k_qkv<<<kNT / 16 / 4, 128, 0, stream>>>(z, g1Wq, g1bq, g1Wk, g1bk,
      g1Wv, g1bv, g1Ws, g1bs, 64, 16, qkv, skip);
  k_zero<<<(accLen + 255) / 256, 256, 0, stream>>>(denom, accLen);
  k_edge_attn<16><<<kET / 256, 256, 0, stream>>>(eidx, eattr, qkv, g1We, denom, num);
  k_finalize<16><<<(kNT * 16) / 256, 256, 0, stream>>>(num, denom, skip, a1);

  // ---- TransformerConv g2 (ch=32, HC=128) ----
  k_qkv<<<kNT / 16 / 4, 128, 0, stream>>>(a1, g2Wq, g2bq, g2Wk, g2bk,
      g2Wv, g2bv, g2Ws, g2bs, 128, 32, qkv, skip);
  k_zero<<<(accLen + 255) / 256, 256, 0, stream>>>(denom, accLen);
  k_edge_attn<32><<<kET / 256, 256, 0, stream>>>(eidx, eattr, qkv, g2We, denom, num);
  k_finalize<32><<<(kNT * 32) / 256, 256, 0, stream>>>(num, denom, skip, a2);

  // ---- relu + agent gather ----
  k_gather<<<(kB * kOUT) / 256, 256, 0, stream>>>(a2, agent, (float*)d_out);
}
